// GraphAttentionLayer_67284957659471
// MI455X (gfx1250) — matrix operand order repaired and checked by reading, not compiled, not run on hardware
//
#include <hip/hip_runtime.h>
#include <cmath>

// ---------------------------------------------------------------------------
// GraphAttentionLayer forward for MI455X (gfx1250, wave32, WMMA bf16).
// All GEMM-shaped work goes through v_wmma_f32_16x16x32_bf16 with fused
// epilogues (alpha, folded-BN row scale/bias, col bias, GELU, residual).
// Staging uses vectorized b128 global->LDS fast paths for full tiles.
// ---------------------------------------------------------------------------

typedef __bf16 bf16;
typedef __attribute__((ext_vector_type(16))) bf16 v16bf;
typedef __attribute__((ext_vector_type(8)))  bf16 v8bf;
typedef __attribute__((ext_vector_type(8)))  float v8f;

static constexpr int kB = 8, kCLS = 20, kC = 256, kN = 3136, kNH = 4, kDH = 64;
static constexpr int kKNN = 9, kCG = 80, k2CG = 160, kGRP = 4;
static constexpr int kTOK = kCLS + kN;          // 3156
static constexpr int kMFF = kB * kTOK;          // 25248
static constexpr int kFF = 1024;
static constexpr int kCH = 392;                 // KNN Gram chunk rows = N/8
static constexpr float kBNS = 0.999995000037499687f;  // 1/sqrt(1+1e-5)

__device__ __forceinline__ float gelu_f(float x) {
  return 0.5f * x * (1.0f + erff(x * 0.7071067811865476f));
}

// ------------------------------ WMMA GEMM ----------------------------------
// C[z] = alpha * A[z] * B[z]  (+ epilogue), bf16 inputs, f32 accumulate.
// flags: 1 = A col-major, 2 = B col-major, 4 = C (and residual) col-major,
//        8 = gelu. Batch offsets: ptr + (z/zdiv)*outer + (z%zdiv)*inner.
struct GemmP {
  const bf16* A; const bf16* Bm; float* Cf; bf16* Cb;
  const float* res; const float* rs; const float* rb; const float* cb;
  long long ao, ai, bo, bi, co, ci, ro, ri;
  int M, N, K, lda, ldb, ldc, ldr, zdiv, rv_mod, flags;
  float alpha;
};

__global__ __launch_bounds__(256) void gemm_bf16_wmma(GemmP p) {
  __shared__ __attribute__((aligned(64))) bf16 As[64][32];
  __shared__ __attribute__((aligned(64))) bf16 Bs[128][32];
  const int tid = threadIdx.x;
  const int z = blockIdx.z;
  const int m0 = blockIdx.y * 64;
  const int n0 = blockIdx.x * 128;
  const int zq = z / p.zdiv, zr = z - zq * p.zdiv;
  const bf16* Ab = p.A  + (long long)zq * p.ao + (long long)zr * p.ai;
  const bf16* Bb = p.Bm + (long long)zq * p.bo + (long long)zr * p.bi;
  const bool a_cm = p.flags & 1, b_cm = p.flags & 2;
  const bool c_cm = p.flags & 4, g_act = p.flags & 8;
  const bool lda8 = (p.lda & 7) == 0, ldb8 = (p.ldb & 7) == 0;
  const bool a_full = (m0 + 64 <= p.M), b_full = (n0 + 128 <= p.N);

  const int wave = tid >> 5, lane = tid & 31;
  const int wm = wave >> 2, wn = wave & 3;       // 2 x 4 wave grid, 32x32/wave
  const int lrow = lane & 15, lhi = lane >> 4;

  v8f acc[2][2];
#pragma unroll
  for (int i = 0; i < 2; ++i)
#pragma unroll
    for (int j = 0; j < 2; ++j)
#pragma unroll
      for (int r = 0; r < 8; ++r) acc[i][j][r] = 0.0f;

  for (int kk = 0; kk < p.K; kk += 32) {
    const bool kfull = (kk + 32 <= p.K);
    // ---------------- stage A tile [64][32] ----------------
    if (kfull && lda8 && a_full) {
      if (!a_cm) {
        int rr = tid >> 2, cc = (tid & 3) << 3;  // 8 contiguous k per thread
        const bf16* src = Ab + (long long)(m0 + rr) * p.lda + (kk + cc);
        *(v8bf*)&As[rr][cc] = *(const v8bf*)src;
        if (kk + 32 < p.K) __builtin_prefetch(src + 32, 0, 1);
      } else {
        int kcol = tid >> 3, mg = (tid & 7) << 3; // 8 contiguous m per thread
        const bf16* src = Ab + (long long)(kk + kcol) * p.lda + (m0 + mg);
        v8bf v = *(const v8bf*)src;
#pragma unroll
        for (int u = 0; u < 8; ++u) As[mg + u][kcol] = v[u];
      }
    } else {
      for (int e = tid; e < 64 * 32; e += 256) {
        int rr = e >> 5, cc = e & 31;
        int gm = m0 + rr, gk = kk + cc;
        bf16 v = (bf16)0.0f;
        if (gm < p.M && gk < p.K)
          v = a_cm ? Ab[(long long)gk * p.lda + gm] : Ab[(long long)gm * p.lda + gk];
        As[rr][cc] = v;
      }
    }
    // ---------------- stage B tile transposed Bs[n][k] ----------------
    if (kfull && ldb8 && b_full) {
      if (!b_cm) {
        int kcol = tid >> 3, ng = (tid & 7) << 4; // 16 contiguous n per thread
        const bf16* src = Bb + (long long)(kk + kcol) * p.ldb + (n0 + ng);
        v8bf v0 = *(const v8bf*)src;
        v8bf v1 = *(const v8bf*)(src + 8);
        if (kk + 32 < p.K) __builtin_prefetch(src + (long long)32 * p.ldb, 0, 1);
#pragma unroll
        for (int u = 0; u < 8; ++u) {
          Bs[ng + u][kcol] = v0[u];
          Bs[ng + 8 + u][kcol] = v1[u];
        }
      } else {
        int nn = tid >> 1, kc = (tid & 1) << 4;   // 16 contiguous k per thread
        const bf16* src = Bb + (long long)(n0 + nn) * p.ldb + (kk + kc);
        *(v8bf*)&Bs[nn][kc]     = *(const v8bf*)src;
        *(v8bf*)&Bs[nn][kc + 8] = *(const v8bf*)(src + 8);
        if (kk + 32 < p.K) __builtin_prefetch(src + 32, 0, 1);
      }
    } else {
      for (int e = tid; e < 128 * 32; e += 256) {
        int nn = e >> 5, cc = e & 31;
        int gn = n0 + nn, gk = kk + cc;
        bf16 v = (bf16)0.0f;
        if (gn < p.N && gk < p.K)
          v = b_cm ? Bb[(long long)gn * p.ldb + gk] : Bb[(long long)gk * p.ldb + gn];
        Bs[nn][cc] = v;
      }
    }
    __syncthreads();
    // Fragments in contiguous k order, same for A and B:
    // A lane l(m=l&15): halves 0-15 -> k=g*16..
    // B lane l(n=l&15): halves 0-15 -> k=g*16..
    v16bf af[2], bfr[2];
#pragma unroll
    for (int i = 0; i < 2; ++i) {
      const v8bf* rp = (const v8bf*)(&As[wm * 32 + i * 16 + lrow][0]);
      v8bf lo = rp[2 * lhi], hi = rp[2 * lhi + 1];
      af[i] = __builtin_shufflevector(lo, hi, 0,1,2,3,4,5,6,7,8,9,10,11,12,13,14,15);
    }
#pragma unroll
    for (int j = 0; j < 2; ++j) {
      const v16bf* rp = (const v16bf*)(&Bs[wn * 32 + j * 16 + lrow][0]);
      bfr[j] = rp[lhi];
    }
#pragma unroll
    for (int i = 0; i < 2; ++i)
#pragma unroll
      for (int j = 0; j < 2; ++j)
        acc[i][j] = __builtin_amdgcn_wmma_f32_16x16x32_bf16(
            false, af[i], false, bfr[j], (short)0, acc[i][j], false, false);
    __syncthreads();
  }

  float* Cf = p.Cf ? p.Cf + (long long)zq * p.co + (long long)zr * p.ci : nullptr;
  bf16*  Cb = p.Cb ? p.Cb + (long long)zq * p.co + (long long)zr * p.ci : nullptr;
  const float* R = p.res ? p.res + (long long)zq * p.ro + (long long)zr * p.ri : nullptr;
#pragma unroll
  for (int i = 0; i < 2; ++i)
#pragma unroll
    for (int j = 0; j < 2; ++j)
#pragma unroll
      for (int r = 0; r < 8; ++r) {
        int m = m0 + wm * 32 + i * 16 + lhi * 8 + r;
        int n = n0 + wn * 32 + j * 16 + lrow;
        if (m >= p.M || n >= p.N) continue;
        float v = acc[i][j][r] * p.alpha;
        int ch = p.rv_mod ? (zr % p.rv_mod) * p.M + m : m;
        if (p.rs) v *= p.rs[ch];
        if (p.rb) v += p.rb[ch];
        if (p.cb) v += p.cb[n];
        if (g_act) v = gelu_f(v);
        long long cix = c_cm ? (long long)n * p.ldc + m : (long long)m * p.ldc + n;
        if (R) {
          long long rix = c_cm ? (long long)n * p.ldr + m : (long long)m * p.ldr + n;
          v += R[rix];
        }
        if (Cf) Cf[cix] = v;
        if (Cb) Cb[cix] = (bf16)v;
      }
}

// --------------------------- elementwise kernels ---------------------------
__global__ void k_f2b(const float* in, bf16* out, long long n) {
  long long t = (long long)blockIdx.x * 256 + threadIdx.x;
  if (t < n) out[t] = (bf16)in[t];
}

// LayerNorm over last dim (C=256), one block per row, writes bf16.
__global__ void k_ln_rows(const float* x, const float* w, const float* bia, bf16* out) {
  __shared__ float sd[256];
  int row = blockIdx.x, tid = threadIdx.x;
  float v = x[(long long)row * kC + tid];
  sd[tid] = v; __syncthreads();
  for (int s = 128; s > 0; s >>= 1) { if (tid < s) sd[tid] += sd[tid + s]; __syncthreads(); }
  float m = sd[0] / kC; __syncthreads();
  float c = v - m;
  sd[tid] = c * c; __syncthreads();
  for (int s = 128; s > 0; s >>= 1) { if (tid < s) sd[tid] += sd[tid + s]; __syncthreads(); }
  float r = rsqrtf(sd[0] / kC + 1e-5f);
  out[(long long)row * kC + tid] = (bf16)(c * r * w[tid] + bia[tid]);
}

// Global LayerNorm over (C,H,W): per-batch stats, then apply->bf16.
__global__ void k_ln_stats(const float* x, float* stats) {
  __shared__ float s1[256], s2[256];
  int b = blockIdx.x, tid = threadIdx.x;
  const float* xp = x + (long long)b * kC * kN;
  float a = 0.f, q = 0.f;
  for (long long i = tid; i < (long long)kC * kN; i += 256) { float v = xp[i]; a += v; q += v * v; }
  s1[tid] = a; s2[tid] = q; __syncthreads();
  for (int s = 128; s > 0; s >>= 1) {
    if (tid < s) { s1[tid] += s1[tid + s]; s2[tid] += s2[tid + s]; }
    __syncthreads();
  }
  if (tid == 0) {
    float cnt = (float)((long long)kC * kN);
    float m = s1[0] / cnt, var = s2[0] / cnt - m * m;
    stats[b * 2] = m; stats[b * 2 + 1] = rsqrtf(var + 1e-5f);
  }
}
__global__ void k_ln_apply(const float* x, const float* stats, const float* w,
                           const float* bia, bf16* out) {
  long long t = (long long)blockIdx.x * 256 + threadIdx.x;
  if (t >= (long long)kB * kC * kN) return;
  int c = (int)((t / kN) % kC);
  int b = (int)(t / ((long long)kC * kN));
  out[t] = (bf16)((x[t] - stats[b * 2]) * stats[b * 2 + 1] * w[c] + bia[c]);
}

// Row softmax over length len (f32 in, bf16 out), one block per row.
__global__ void k_softmax_row(const float* in, bf16* out, int len) {
  __shared__ float sd[256];
  int row = blockIdx.x, tid = threadIdx.x;
  const float* ip = in + (long long)row * len;
  bf16* op = out + (long long)row * len;
  float mx = -3.4e38f;
  for (int i = tid; i < len; i += 256) mx = fmaxf(mx, ip[i]);
  sd[tid] = mx; __syncthreads();
  for (int s = 128; s > 0; s >>= 1) { if (tid < s) sd[tid] = fmaxf(sd[tid], sd[tid + s]); __syncthreads(); }
  mx = sd[0]; __syncthreads();
  float sum = 0.f;
  for (int i = tid; i < len; i += 256) sum += __expf(ip[i] - mx);
  sd[tid] = sum; __syncthreads();
  for (int s = 128; s > 0; s >>= 1) { if (tid < s) sd[tid] += sd[tid + s]; __syncthreads(); }
  float inv = 1.0f / sd[0];
  for (int i = tid; i < len; i += 256) op[i] = (bf16)(__expf(ip[i] - mx) * inv);
}

// 3x3 conv (B,20,56,56)->(B,256,56,56) fused +bias, BN-affine, GELU.
__global__ void k_conv3x3(const float* in, const float* w, const float* cb,
                          const float* g, const float* bb, float* out) {
  long long t = (long long)blockIdx.x * 256 + threadIdx.x;
  if (t >= (long long)kB * kC * kN) return;
  int x = (int)(t % 56), y = (int)((t / 56) % 56);
  int o = (int)((t / kN) % kC), b = (int)(t / ((long long)kC * kN));
  const float* wp = w + (long long)o * kCLS * 9;
  const float* ip = in + (long long)b * kCLS * kN;
  float acc = 0.f;
  for (int i = 0; i < kCLS; ++i)
    for (int dy = 0; dy < 3; ++dy) {
      int yy = y + dy - 1; if (yy < 0 || yy >= 56) continue;
      for (int dx = 0; dx < 3; ++dx) {
        int xx = x + dx - 1; if (xx < 0 || xx >= 56) continue;
        acc += ip[(i * 56 + yy) * 56 + xx] * wp[i * 9 + dy * 3 + dx];
      }
    }
  float v = (acc + cb[o]) * (g[o] * kBNS) + bb[o];
  out[t] = gelu_f(v);
}

// kv2 (B,20,512) -> key2 (B,4,64,20) and val2 (B,4,20,64), bf16.
__global__ void k_split_kv2(const float* kv2, bf16* key2, bf16* val2) {
  int t = blockIdx.x * 256 + threadIdx.x;
  if (t >= kB * kCLS * 2 * kC) return;
  int cch = t % (2 * kC), cls = (t / (2 * kC)) % kCLS, b = t / (2 * kC * kCLS);
  float v = kv2[t];
  if (cch < kC) {
    int h = cch / kDH, d = cch % kDH;
    key2[(((long long)b * kNH + h) * kDH + d) * kCLS + cls] = (bf16)v;
  } else {
    int c2 = cch - kC; int h = c2 / kDH, d = c2 % kDH;
    val2[(((long long)b * kNH + h) * kCLS + cls) * kDH + d] = (bf16)v;
  }
}

// Normalize grapher features: y (B,80,N) -> fnT (B,80,N) bf16, fn (B,N,80) bf16, sq (B,N).
__global__ void k_fnorm(const float* y, bf16* fnT, bf16* fn, float* sq) {
  int t = blockIdx.x * 256 + threadIdx.x;
  if (t >= kB * kN) return;
  int n = t % kN, b = t / kN;
  const float* yp = y + (long long)b * kCG * kN + n;
  float s = 0.f;
  for (int c = 0; c < kCG; ++c) { float v = yp[(long long)c * kN]; s += v * v; }
  float inv = 1.0f / fmaxf(sqrtf(s), 1e-12f);
  sq[t] = s * inv * inv;
  bf16* fp = fn + (long long)t * kCG;
  for (int c = 0; c < kCG; ++c) {
    float v = yp[(long long)c * kN] * inv;
    fnT[(long long)b * kCG * kN + (long long)c * kN + n] = (bf16)v;
    fp[c] = (bf16)v;
  }
}

// Top-9 of neg_dist = 2*G - sq[n] - sq[m] over a Gram chunk.
__global__ void k_topk(const float* G, const float* sq, int* idx, int chunk_start) {
  int t = blockIdx.x * 128 + threadIdx.x;
  if (t >= kB * kCH) return;
  int r = t % kCH, b = t / kCH;
  const float* gp = G + ((long long)b * kCH + r) * kN;
  const float* sp = sq + (long long)b * kN;
  int n = chunk_start + r;
  float sn = sp[n];
  float bv[kKNN]; int bi[kKNN];
  for (int j = 0; j < kKNN; ++j) { bv[j] = -3.4e38f; bi[j] = 0; }
  for (int m = 0; m < kN; ++m) {
    float v = 2.0f * gp[m] - sn - sp[m];
    if (v > bv[kKNN - 1]) {
      int pos = kKNN - 1;
      while (pos > 0 && v > bv[pos - 1]) { bv[pos] = bv[pos - 1]; bi[pos] = bi[pos - 1]; --pos; }
      bv[pos] = v; bi[pos] = m;
    }
  }
  int* op = idx + ((long long)b * kN + n) * kKNN;
  for (int j = 0; j < kKNN; ++j) op[j] = bi[j];
}

// Build z (B,160,N) bf16: z[2c]=feat, z[2c+1]=max_j(feat[idx_j]-feat).
__global__ void k_build_z(const float* y, const int* idx, bf16* z) {
  long long t = (long long)blockIdx.x * 256 + threadIdx.x;
  if (t >= (long long)kB * kN * kCG) return;
  int c = (int)(t % kCG), n = (int)((t / kCG) % kN), b = (int)(t / ((long long)kCG * kN));
  const float* yb = y + (long long)b * kCG * kN;
  float f = yb[(long long)c * kN + n];
  const int* ip = idx + ((long long)b * kN + n) * kKNN;
  float best = -3.4e38f;
  for (int j = 0; j < kKNN; ++j) best = fmaxf(best, yb[(long long)c * kN + ip[j]] - f);
  bf16* zb = z + (long long)b * k2CG * kN;
  zb[(long long)(2 * c) * kN + n] = (bf16)f;
  zb[(long long)(2 * c + 1) * kN + n] = (bf16)best;
}

// Softmax over CLS (strided read from (B,80,N)) -> asm (B,4,N,20) bf16.
__global__ void k_softmax_cls(const float* a, bf16* out) {
  int t = blockIdx.x * 256 + threadIdx.x;
  if (t >= kB * kNH * kN) return;
  int n = t % kN, h = (t / kN) % kNH, b = t / (kN * kNH);
  const float* ap = a + ((long long)b * kCG + h * kCLS) * kN + n;
  float v[kCLS]; float mx = -3.4e38f;
  for (int c = 0; c < kCLS; ++c) { v[c] = ap[(long long)c * kN]; mx = fmaxf(mx, v[c]); }
  float s = 0.f;
  for (int c = 0; c < kCLS; ++c) { v[c] = __expf(v[c] - mx); s += v[c]; }
  float inv = 1.0f / s;
  bf16* op = out + (((long long)(b * kNH + h)) * kN + n) * kCLS;
  for (int c = 0; c < kCLS; ++c) op[c] = (bf16)(v[c] * inv);
}

// Concatenate [x_cls (B,20,256) ; x_patch^T (B,N,256)] -> xcat bf16 (MFF,256).
__global__ void k_build_xcat(const float* xcls, const float* xp, bf16* out) {
  long long t = (long long)blockIdx.x * 256 + threadIdx.x;
  if (t >= (long long)kMFF * kC) return;
  int cidx = (int)(t % kC);
  long long row = t / kC;
  int tok = (int)(row % kTOK), b = (int)(row / kTOK);
  float v;
  if (tok < kCLS) v = xcls[((long long)b * kCLS + tok) * kC + cidx];
  else            v = xp[((long long)b * kC + cidx) * kN + (tok - kCLS)];
  out[t] = (bf16)v;
}

// Scatter FFN output into d_out: x_cls flat then x_patch (B,C,H,W).
__global__ void k_final(const float* xout, float* dout) {
  long long t = (long long)blockIdx.x * 256 + threadIdx.x;
  long long clsN = (long long)kB * kCLS * kC;
  long long total = clsN + (long long)kB * kC * kN;
  if (t >= total) return;
  if (t < clsN) {
    int cidx = (int)(t % kC);
    long long row = t / kC;
    int tok = (int)(row % kCLS), b = (int)(row / kCLS);
    dout[t] = xout[((long long)b * kTOK + tok) * kC + cidx];
  } else {
    long long u = t - clsN;
    int n = (int)(u % kN);
    long long r2 = u / kN;
    int cidx = (int)(r2 % kC), b = (int)(r2 / kC);
    dout[t] = xout[((long long)b * kTOK + kCLS + n) * kC + cidx];
  }
}

// Fold BN (and conv bias) into per-channel scale/bias vectors for GEMM epilogues.
__global__ void k_prep_vec(const float* fc1_b, const float* fc1_g, const float* fc1_bb,
                           const float* nn_b,  const float* nn_g,  const float* nn_bb,
                           const float* fc2_b, const float* fc2_g, const float* fc2_bb,
                           float* s1, float* t1, float* s2, float* t2, float* s3, float* t3) {
  int i = threadIdx.x;
  if (i < kCG)  { float s = fc1_g[i] * kBNS; s1[i] = s; t1[i] = fc1_b[i] * s + fc1_bb[i]; }
  if (i < k2CG) { float s = nn_g[i]  * kBNS; s2[i] = s; t2[i] = nn_b[i]  * s + nn_bb[i]; }
  if (i < kCG)  { float s = fc2_g[i] * kBNS; s3[i] = s; t3[i] = fc2_b[i] * s + fc2_bb[i]; }
}

// ------------------------------- host side ---------------------------------
static void gemm(hipStream_t st, int batches, GemmP p) {
  dim3 g((p.N + 127) / 128, (p.M + 63) / 64, batches), blk(256, 1, 1);
  gemm_bf16_wmma<<<g, blk, 0, st>>>(p);
}
static inline GemmP GP() { GemmP p{}; p.zdiv = 1; p.alpha = 1.0f; return p; }

extern "C" void kernel_launch(void* const* d_in, const int* in_sizes, int n_in,
                              void* d_out, int out_size, void* d_ws, size_t ws_size,
                              hipStream_t stream) {
  (void)in_sizes; (void)n_in; (void)out_size; (void)ws_size;
  auto F = [&](int i) { return (const float*)d_in[i]; };
  const float* x_cls = F(0);
  const float* x_patch = F(1);

  // ---- workspace bump allocator ----
  char* base = (char*)d_ws;
  size_t off = 0;
  auto alloc = [&](size_t bytes) -> char* {
    char* p = base + off; off += (bytes + 255) & ~(size_t)255; return p;
  };
  auto fb = [&](long long n) { return (float*)alloc((size_t)n * 4); };
  auto hb = [&](long long n) { return (bf16*)alloc((size_t)n * 2); };

  const long long CN = (long long)kC * kN;      // 802816
  // bf16 weights
  bf16 *w_q = hb(kC * kC),    *w_kv = hb(2 * kC * kC), *w_pr = hb(kC * kC);
  bf16 *w_q2 = hb(kC * kC),   *w_kv2 = hb(kC * 2 * kC), *w_pr2 = hb(kC * kC);
  bf16 *w_g1 = hb(kCG * kCG), *w_gnn = hb(k2CG * 40),   *w_g2 = hb(kCG * k2CG);
  bf16 *w_f1 = hb(kC * kFF),  *w_f2 = hb(kFF * kC);
  // activations
  bf16* xpatch_bf = hb(kB * CN);
  bf16* xcls_ln = hb(kB * kCLS * kC);
  bf16* qcls_bf = hb(kB * kCLS * kC);
  bf16* kvconv_bf = hb((long long)kB * 2 * CN);
  float* attn_f = fb((long long)kB * kNH * kCLS * kN);
  bf16* attnsm_bf = hb((long long)kB * kNH * kCLS * kN);
  bf16* outcat_bf = hb(kB * kCLS * kC);
  float* outcls2_f = fb(kB * kCLS * kC);
  bf16* outcls2_bf = hb(kB * kCLS * kC);
  float* op20_f = fb((long long)kB * kCLS * kN);
  float* xp2_f = fb(kB * CN);
  float* stats = fb(2 * kB);
  bf16* xnorm_bf = hb(kB * CN);                 // reused for both global LNs
  bf16* qout2_bf = hb(kB * CN);
  float* kv2_f = fb(kB * kCLS * 2 * kC);
  bf16* key2_bf = hb(kB * kNH * kDH * kCLS);
  bf16* val2_bf = hb(kB * kNH * kCLS * kDH);
  float* a_f = fb((long long)kB * kCG * kN);
  bf16* a_bf = hb((long long)kB * kCG * kN);
  float* y_f = fb((long long)kB * kCG * kN);
  bf16* fnT_bf = hb((long long)kB * kCG * kN);
  bf16* fn_bf = hb((long long)kB * kN * kCG);
  float* sq_f = fb(kB * kN);
  int* idx_i = (int*)alloc((size_t)kB * kN * kKNN * 4);
  // Gram chunk region, later reused as FFN hidden (52MB > 39MB)
  char* shreg = alloc((size_t)kMFF * kFF * 2);
  float* Gbuf = (float*)shreg;
  bf16* hbuf_bf = (bf16*)shreg;
  bf16* z_bf = hb((long long)kB * k2CG * kN);
  bf16* z2_bf = hb((long long)kB * k2CG * kN);
  float* anew_f = fb((long long)kB * kCG * kN);
  bf16* asm_bf = hb((long long)kB * kNH * kN * kCLS);
  bf16* outT_bf = hb(kB * CN);
  float* xp3_f = fb(kB * CN);
  bf16* xcat_bf = hb((long long)kMFF * kC);
  float* xout_f = fb((long long)kMFF * kC);
  float *s1 = fb(kCG), *t1 = fb(kCG), *s2 = fb(k2CG), *t2 = fb(k2CG), *s3 = fb(kCG), *t3 = fb(kCG);

  auto conv = [&](const float* src, bf16* dst, long long n) {
    k_f2b<<<(unsigned)((n + 255) / 256), 256, 0, stream>>>(src, dst, n);
  };
  // ---- weight conversions ----
  conv(F(4), w_q, kC * kC);       conv(F(8), w_kv, 2 * kC * kC);
  conv(F(10), w_pr, kC * kC);     conv(F(18), w_q2, kC * kC);
  conv(F(22), w_kv2, kC * 2 * kC); conv(F(24), w_pr2, kC * kC);
  conv(F(26), w_g1, kCG * kCG);   conv(F(30), w_gnn, k2CG * 40);
  conv(F(34), w_g2, kCG * k2CG);  conv(F(38), w_f1, kC * kFF);
  conv(F(40), w_f2, kFF * kC);
  conv(x_patch, xpatch_bf, kB * CN);
  k_prep_vec<<<1, 256, 0, stream>>>(F(27), F(28), F(29), F(31), F(32), F(33),
                                    F(35), F(36), F(37), s1, t1, s2, t2, s3, t3);

  // ================= sem2feat =================
  k_ln_rows<<<kB * kCLS, 256, 0, stream>>>(x_cls, F(2), F(3), xcls_ln);
  k_ln_stats<<<kB, 256, 0, stream>>>(x_patch, stats);
  k_ln_apply<<<(unsigned)((kB * CN + 255) / 256), 256, 0, stream>>>(x_patch, stats, F(6), F(7), xnorm_bf);

  { // q = ln(x_cls) @ Wq + b  -> bf16 (160,256)
    GemmP p = GP(); p.A = xcls_ln; p.Bm = w_q; p.Cb = qcls_bf; p.cb = F(5);
    p.M = kB * kCLS; p.N = kC; p.K = kC; p.lda = kC; p.ldb = kC; p.ldc = kC;
    gemm(stream, 1, p);
  }
  { // kv = Wkv(512,256) * xnorm(256,N) + b  -> bf16 (B,512,N)
    GemmP p = GP(); p.A = w_kv; p.Bm = xnorm_bf; p.Cb = kvconv_bf; p.rb = F(9);
    p.M = 2 * kC; p.N = kN; p.K = kC; p.lda = kC; p.ldb = kN; p.ldc = kN;
    p.bo = CN; p.co = 2 * CN;
    gemm(stream, kB, p);
  }
  { // attn = 0.125 * q(20,64) @ key(64,N) per (b,h)
    GemmP p = GP(); p.zdiv = kNH; p.alpha = 0.125f;
    p.A = qcls_bf; p.lda = kC; p.ao = (long long)kCLS * kC; p.ai = kDH;
    p.Bm = kvconv_bf; p.ldb = kN; p.bo = 2 * CN; p.bi = (long long)kDH * kN;
    p.Cf = attn_f; p.ldc = kN; p.co = (long long)kNH * kCLS * kN; p.ci = (long long)kCLS * kN;
    p.M = kCLS; p.N = kN; p.K = kDH;
    gemm(stream, kB * kNH, p);
  }
  k_softmax_row<<<kB * kNH * kCLS, 256, 0, stream>>>(attn_f, attnsm_bf, kN);
  { // out = attn_sm(20,N) @ val(N,64), val = kvconv[:,256+...] col-major
    GemmP p = GP(); p.zdiv = kNH; p.flags = 2;
    p.A = attnsm_bf; p.lda = kN; p.ao = (long long)kNH * kCLS * kN; p.ai = (long long)kCLS * kN;
    p.Bm = kvconv_bf + (long long)kC * kN; p.ldb = kN; p.bo = 2 * CN; p.bi = (long long)kDH * kN;
    p.Cb = outcat_bf; p.ldc = kC; p.co = (long long)kCLS * kC; p.ci = kDH;
    p.M = kCLS; p.N = kDH; p.K = kN;
    gemm(stream, kB * kNH, p);
  }
  { // out_cls = x_cls + out @ Wproj + b
    GemmP p = GP(); p.A = outcat_bf; p.Bm = w_pr; p.Cf = outcls2_f; p.Cb = outcls2_bf;
    p.cb = F(11); p.res = x_cls; p.ldr = kC;
    p.M = kB * kCLS; p.N = kC; p.K = kC; p.lda = kC; p.ldb = kC; p.ldc = kC;
    gemm(stream, 1, p);
  }
  { // out_patch20 = out_cls(20,256) @ x_patch(256,N) per b
    GemmP p = GP(); p.A = outcls2_bf; p.lda = kC; p.ao = (long long)kCLS * kC;
    p.Bm = xpatch_bf; p.ldb = kN; p.bo = CN;
    p.Cf = op20_f; p.ldc = kN; p.co = (long long)kCLS * kN;
    p.M = kCLS; p.N = kN; p.K = kC;
    gemm(stream, kB, p);
  }
  k_conv3x3<<<(unsigned)((kB * CN + 255) / 256), 256, 0, stream>>>(op20_f, F(12), F(13), F(14), F(15), xp2_f);

  // ================= feat2sem =================
  k_ln_stats<<<kB, 256, 0, stream>>>(xp2_f, stats);
  k_ln_apply<<<(unsigned)((kB * CN + 255) / 256), 256, 0, stream>>>(xp2_f, stats, F(16), F(17), xnorm_bf);
  { // q2 = Wq2(256,256) * xnorm(256,N) + b -> bf16 (B,256,N)
    GemmP p = GP(); p.A = w_q2; p.Bm = xnorm_bf; p.Cb = qout2_bf; p.rb = F(19);
    p.M = kC; p.N = kN; p.K = kC; p.lda = kC; p.ldb = kN; p.ldc = kN;
    p.bo = CN; p.co = CN;
    gemm(stream, kB, p);
  }
  k_ln_rows<<<kB * kCLS, 256, 0, stream>>>(outcls2_f, F(20), F(21), xcls_ln);
  { // kv2 = ln(out_cls) @ Wkv2(256,512) + b
    GemmP p = GP(); p.A = xcls_ln; p.Bm = w_kv2; p.Cf = kv2_f; p.cb = F(23);
    p.M = kB * kCLS; p.N = 2 * kC; p.K = kC; p.lda = kC; p.ldb = 2 * kC; p.ldc = 2 * kC;
    gemm(stream, 1, p);
  }
  k_split_kv2<<<(kB * kCLS * 2 * kC + 255) / 256, 256, 0, stream>>>(kv2_f, key2_bf, val2_bf);
  { // a = 0.125 * q2^T(N,64) @ key2(64,20), stored col-major -> (B,80,N)
    GemmP p = GP(); p.zdiv = kNH; p.flags = 1 | 4; p.alpha = 0.125f;
    p.A = qout2_bf; p.lda = kN; p.ao = CN; p.ai = (long long)kDH * kN;
    p.Bm = key2_bf; p.ldb = kCLS; p.bo = (long long)kNH * kDH * kCLS; p.bi = (long long)kDH * kCLS;
    p.Cf = a_f; p.Cb = a_bf; p.ldc = kN; p.co = (long long)kCG * kN; p.ci = (long long)kCLS * kN;
    p.M = kN; p.N = kCLS; p.K = kDH;
    gemm(stream, kB * kNH, p);
  }
  // ---- grapher ----
  { // y = bn(W_fc1(80,80) * a(80,N) + b)
    GemmP p = GP(); p.A = w_g1; p.Bm = a_bf; p.Cf = y_f; p.rs = s1; p.rb = t1;
    p.M = kCG; p.N = kN; p.K = kCG; p.lda = kCG; p.ldb = kN; p.ldc = kN;
    p.bo = (long long)kCG * kN; p.co = (long long)kCG * kN;
    gemm(stream, kB, p);
  }
  k_fnorm<<<(kB * kN + 255) / 256, 256, 0, stream>>>(y_f, fnT_bf, fn_bf, sq_f);
  for (int c = 0; c < kN / kCH; ++c) {  // chunked Gram + top-k
    GemmP p = GP();
    p.A = fn_bf + (long long)c * kCH * kCG; p.lda = kCG; p.ao = (long long)kN * kCG;
    p.Bm = fnT_bf; p.ldb = kN; p.bo = (long long)kCG * kN;
    p.Cf = Gbuf; p.ldc = kN; p.co = (long long)kCH * kN;
    p.M = kCH; p.N = kN; p.K = kCG;
    gemm(stream, kB, p);
    k_topk<<<(kB * kCH + 127) / 128, 128, 0, stream>>>(Gbuf, sq_f, idx_i, c * kCH);
  }
  k_build_z<<<(unsigned)(((long long)kB * kN * kCG + 255) / 256), 256, 0, stream>>>(y_f, idx_i, z_bf);
  { // grouped nn: per (b,g): gelu(bn(Wg(40,40) * z_g(40,N) + b))
    GemmP p = GP(); p.zdiv = kGRP; p.flags = 8; p.rv_mod = kGRP;
    p.A = w_gnn; p.lda = 40; p.ai = 40 * 40;
    p.Bm = z_bf; p.ldb = kN; p.bo = (long long)k2CG * kN; p.bi = (long long)40 * kN;
    p.Cb = z2_bf; p.ldc = kN; p.co = (long long)k2CG * kN; p.ci = (long long)40 * kN;
    p.rs = s2; p.rb = t2;
    p.M = 40; p.N = kN; p.K = 40;
    gemm(stream, kB * kGRP, p);
  }
  { // a_new = bn(W_fc2(80,160) * z2(160,N) + b) + a
    GemmP p = GP(); p.A = w_g2; p.Bm = z2_bf; p.Cf = anew_f; p.rs = s3; p.rb = t3;
    p.res = a_f; p.ldr = kN; p.ro = (long long)kCG * kN;
    p.M = kCG; p.N = kN; p.K = k2CG; p.lda = k2CG; p.ldb = kN; p.ldc = kN;
    p.bo = (long long)k2CG * kN; p.co = (long long)kCG * kN;
    gemm(stream, kB, p);
  }
  k_softmax_cls<<<(kB * kNH * kN + 255) / 256, 256, 0, stream>>>(anew_f, asm_bf);
  { // out = asm(N,20) @ val2(20,64), stored col-major -> (B,256,N) bf16
    GemmP p = GP(); p.zdiv = kNH; p.flags = 4;
    p.A = asm_bf; p.lda = kCLS; p.ao = (long long)kNH * kN * kCLS; p.ai = (long long)kN * kCLS;
    p.Bm = val2_bf; p.ldb = kDH; p.bo = (long long)kNH * kCLS * kDH; p.bi = (long long)kCLS * kDH;
    p.Cb = outT_bf; p.ldc = kN; p.co = CN; p.ci = (long long)kDH * kN;
    p.M = kN; p.N = kDH; p.K = kCLS;
    gemm(stream, kB * kNH, p);
  }
  { // x_patch3 = xp2 + Wproj2(256,256) * out(256,N) + b
    GemmP p = GP(); p.A = w_pr2; p.Bm = outT_bf; p.Cf = xp3_f; p.rb = F(25);
    p.res = xp2_f; p.ldr = kN; p.ro = CN;
    p.M = kC; p.N = kN; p.K = kC; p.lda = kC; p.ldb = kN; p.ldc = kN;
    p.bo = CN; p.co = CN;
    gemm(stream, kB, p);
  }
  // ================= FFN =================
  k_build_xcat<<<(unsigned)(((long long)kMFF * kC + 255) / 256), 256, 0, stream>>>(outcls2_f, xp3_f, xcat_bf);
  { // h = gelu(x @ W1 + b1)
    GemmP p = GP(); p.flags = 8; p.A = xcat_bf; p.Bm = w_f1; p.Cb = hbuf_bf; p.cb = F(39);
    p.M = kMFF; p.N = kFF; p.K = kC; p.lda = kC; p.ldb = kFF; p.ldc = kFF;
    gemm(stream, 1, p);
  }
  { // xout = h @ W2 + b2
    GemmP p = GP(); p.A = hbuf_bf; p.Bm = w_f2; p.Cf = xout_f; p.cb = F(41);
    p.M = kMFF; p.N = kC; p.K = kFF; p.lda = kFF; p.ldb = kC; p.ldc = kC;
    gemm(stream, 1, p);
  }
  long long tot = (long long)kB * kCLS * kC + (long long)kB * kC * kN;
  k_final<<<(unsigned)((tot + 255) / 256), 256, 0, stream>>>(xout_f, (float*)d_out);
}
